// FullMultiHeadConvAttention_2808908612204
// MI455X (gfx1250) — matrix operand order repaired and checked by reading, not compiled, not run on hardware
//
#include <hip/hip_runtime.h>
#include <hip/hip_bf16.h>
#include <stdint.h>
#include <stddef.h>

// ---------------------------------------------------------------------------
// MI455X (gfx1250): bf16 WMMA for all GEMMs; TDM (tensor_load_to_lds) +
// double-buffered LDS staging for the two big projections; flash attention
// with hoisted fragment loads. Wave32 fragment layouts per CDNA5 ISA 7.12.2.
// ---------------------------------------------------------------------------

typedef __bf16 bf16_t;
typedef bf16_t bf16x16 __attribute__((ext_vector_type(16)));
typedef float  f32x8   __attribute__((ext_vector_type(8)));
typedef unsigned int u32x4 __attribute__((ext_vector_type(4)));
typedef int    i32x8  __attribute__((ext_vector_type(8)));
typedef int    i32x4  __attribute__((ext_vector_type(4)));

#define D_MODEL  2048
#define N_HEAD   16
#define HEAD_DIM 128
#define SEQ      2048
#define BATCH    2
#define ROWS     (BATCH * SEQ)     // 4096
#define QKV_N    (4 * D_MODEL)     // 8192

#define BM 128
#define BN 128
#define BK 32

struct alignas(16) B128c { bf16_t h[8]; };   // 16B chunk of 8 bf16
struct alignas(8)  B64c  { bf16_t h[4]; };   // 8B chunk of 4 bf16

// ---------------------------------------------------------------------------
// TDM: 2-D tile load global -> LDS. D# packed per CDNA5 ISA §8.3/8.4.
//   tile_d0 = contiguous elements per row, tile_d1 = rows,
//   row_stride = elements between rows. data_size = 2 bytes (bf16).
// Tensor dims are set equal to tile dims (tiles are always fully in-bounds).
// This toolchain exposes the 6-arg builtin: (g0, g1, g2, g3, g4, cpol).
// ---------------------------------------------------------------------------
__device__ __forceinline__ void tdm_load_2d(uint32_t lds_off, const void* gptr,
                                            uint32_t tile_d0, uint32_t tile_d1,
                                            uint32_t row_stride) {
  const uint64_t ga = (uint64_t)(uintptr_t)gptr;
  u32x4 g0;
  g0[0] = 1u;                                                 // count=1 (valid)
  g0[1] = lds_off;                                            // lds_addr
  g0[2] = (uint32_t)ga;                                       // global_addr lo
  g0[3] = (uint32_t)((ga >> 32) & 0x01FFFFFFu) | (2u << 30);  // ga[56:32]|type=2
  i32x8 g1;
  g1[0] = 0x00010000;                            // wg_mask=0, data_size=1 (2B)
  g1[1] = (int)((tile_d0 & 0xFFFFu) << 16);      // tensor_dim0[15:0]
  g1[2] = (int)((tile_d0 >> 16) | ((tile_d1 & 0xFFFFu) << 16));
  g1[3] = (int)((tile_d1 >> 16) | (tile_d0 << 16));   // | tile_dim0
  g1[4] = (int)(tile_d1 & 0xFFFFu);              // tile_dim1 (tile_dim2 = 0)
  g1[5] = (int)row_stride;                       // tensor_dim0_stride lo
  g1[6] = 0;
  g1[7] = 0;
  const i32x4 gz4 = {};
  const i32x8 gz8 = {};
  __builtin_amdgcn_tensor_load_to_lds(g0, g1, gz4, gz4, gz8, 0);
}

// ---- A fragment from LDS/global: 16x32 bf16, row-major, ld = row stride ----
// lanes 0-15: M=lane, elems 0-7 -> K=0..7, 8-15 -> K=16..23
// lanes 16-31: M=lane-16, elems 0-7 -> K=8..15, 8-15 -> K=24..31
__device__ __forceinline__ bf16x16 frag_a(const bf16_t* p /* row0/k0 applied */,
                                          int ld, int m, int kb) {
  const bf16_t* q = p + (size_t)m * ld + kb;
  B128c c0 = *(const B128c*)(q);
  B128c c1 = *(const B128c*)(q + 16);
  bf16x16 a;
#pragma unroll
  for (int i = 0; i < 8; ++i) { a[i] = c0.h[i]; a[i + 8] = c1.h[i]; }
  return a;
}

// ---------------------------------------------------------------------------
// fp32 -> bf16 conversion
// ---------------------------------------------------------------------------
__global__ void f32_to_bf16(const float* __restrict__ src,
                            bf16_t* __restrict__ dst, int n) {
  int i = blockIdx.x * blockDim.x + threadIdx.x;
  if (i < n) dst[i] = (bf16_t)src[i];
}

// ---------------------------------------------------------------------------
// GEMM: C[M,N] = A[M,K] x B[K,N] (bf16 in, f32 acc).
// 256 threads = 8 waves; 128x128 workgroup tile; K-panels of 32 staged to LDS
// by the Tensor Data Mover, double buffered (TDM of panel i+1 overlaps the
// 8x4x2 WMMA burst on panel i). Wave w: rows [64*(w>>2)), cols [32*(w&3)).
// ---------------------------------------------------------------------------
template <bool STORE_F32>
__global__ void __launch_bounds__(256)
gemm_bf16_wmma_tdm(const bf16_t* __restrict__ A, const bf16_t* __restrict__ B,
                   void* __restrict__ C, int M, int N, int K) {
  (void)M;
  __shared__ bf16_t lA[2][BM * BK];   // 8 KB per buffer
  __shared__ bf16_t lB[2][BK * BN];   // 8 KB per buffer
  const int tid  = threadIdx.x;
  const int lane = tid & 31;
  const int w    = tid >> 5;
  const int wm   = w >> 2;            // 0..1  -> 64-row band
  const int wn   = w & 3;             // 0..3  -> 32-col band
  const int row0 = blockIdx.y * BM;
  const int col0 = blockIdx.x * BN;

  const int n   = lane & 15;
  const int hfl = lane >> 4;
  const int kbA = hfl ? 8 : 0;        // A-frag K base
  const int koB = hfl ? 8 : 0;        // B-frag K base

  f32x8 acc[4][2];
  const f32x8 zero = {};
#pragma unroll
  for (int i = 0; i < 4; ++i)
#pragma unroll
    for (int j = 0; j < 2; ++j) acc[i][j] = zero;

  // prologue: stage panel 0 into buffer 0
  if (w == 0) {
    tdm_load_2d((uint32_t)(uintptr_t)&lA[0][0], A + (size_t)row0 * K, BK, BM,
                (uint32_t)K);
    tdm_load_2d((uint32_t)(uintptr_t)&lB[0][0], B + col0, BN, BK, (uint32_t)N);
  }

  int cur = 0;
  for (int k0 = 0; k0 < K; k0 += BK) {
    if (w == 0) __builtin_amdgcn_s_wait_tensorcnt(0);
    __syncthreads();  // buf[cur] ready; everyone done reading buf[cur^1]
    if (w == 0 && k0 + BK < K) {
      const int kn = k0 + BK;
      tdm_load_2d((uint32_t)(uintptr_t)&lA[cur ^ 1][0],
                  A + (size_t)row0 * K + kn, BK, BM, (uint32_t)K);
      tdm_load_2d((uint32_t)(uintptr_t)&lB[cur ^ 1][0],
                  B + (size_t)kn * N + col0, BN, BK, (uint32_t)N);
    }
    // B fragments for this wave's two 16-col tiles (LDS, bank-parallel)
    bf16x16 bfr[2];
#pragma unroll
    for (int j = 0; j < 2; ++j) {
      const bf16_t* p = &lB[cur][0] + (size_t)koB * BN + wn * 32 + j * 16 + n;
      bf16x16 bb;
#pragma unroll
      for (int i = 0; i < 16; ++i) bb[i] = p[(i < 8 ? i : i + 8) * BN];
      bfr[j] = bb;
    }
    // 4 row tiles x 2 col tiles of WMMA
#pragma unroll
    for (int i = 0; i < 4; ++i) {
      bf16x16 aa = frag_a(&lA[cur][0] + (size_t)(wm * 64 + i * 16) * BK, BK, n,
                          kbA);
#pragma unroll
      for (int j = 0; j < 2; ++j)
        acc[i][j] = __builtin_amdgcn_wmma_f32_16x16x32_bf16(
            false, aa, false, bfr[j], (short)0, acc[i][j], false, false);
    }
    __syncthreads();  // reads of buf[cur] done before it is refilled
    cur ^= 1;
  }

  // epilogue
  const int mb = hfl ? 8 : 0;
#pragma unroll
  for (int i = 0; i < 4; ++i)
#pragma unroll
    for (int j = 0; j < 2; ++j)
#pragma unroll
      for (int r = 0; r < 8; ++r) {
        const size_t idx =
            (size_t)(row0 + wm * 64 + i * 16 + mb + r) * N +
            col0 + wn * 32 + j * 16 + n;
        if (STORE_F32) ((float*)C)[idx]  = acc[i][j][r];
        else           ((bf16_t*)C)[idx] = (bf16_t)acc[i][j][r];
      }
}

// ---------------------------------------------------------------------------
// Fused depthwise causal conv(3) + RMSNorm + RoPE + sigmoid gate.
// One wave per (b, t, h); lane owns 4 contiguous head dims.
// ---------------------------------------------------------------------------
__global__ void __launch_bounds__(256)
conv_norm_rope(const bf16_t* __restrict__ qkv,
               const float* __restrict__ wq, const float* __restrict__ wk,
               const float* __restrict__ wv, const float* __restrict__ wi,
               const float* __restrict__ qnw, const float* __restrict__ knw,
               const float* __restrict__ fcos, const float* __restrict__ fsin,
               bf16_t* __restrict__ qh, bf16_t* __restrict__ kh,
               bf16_t* __restrict__ vh, float* __restrict__ gate) {
  const int gtid = blockIdx.x * blockDim.x + threadIdx.x;
  const int wid  = gtid >> 5;
  const int lane = gtid & 31;
  const int h = wid & (N_HEAD - 1);
  const int t = (wid >> 4) & (SEQ - 1);
  const int b = wid >> 15;
  const int d0 = lane * 4;
  const int c0 = h * HEAD_DIM + d0;

  float xv[4][3][4];
#pragma unroll
  for (int s = 0; s < 4; ++s) {
    const int colbase = s * D_MODEL + c0;
#pragma unroll
    for (int dt = 0; dt < 3; ++dt) {
      const int tt = t - 2 + dt;
      if (tt >= 0) {
        const bf16_t* p = qkv + (size_t)(b * SEQ + tt) * QKV_N + colbase;
        B64c v4 = *(const B64c*)p;
#pragma unroll
        for (int j = 0; j < 4; ++j) xv[s][dt][j] = (float)v4.h[j];
      } else {
#pragma unroll
        for (int j = 0; j < 4; ++j) xv[s][dt][j] = 0.f;
      }
    }
  }
  float q[4], k[4], v[4], g[4];
#pragma unroll
  for (int j = 0; j < 4; ++j) {
    const int c = c0 + j;
    q[j] = wq[c*3+0]*xv[0][0][j] + wq[c*3+1]*xv[0][1][j] + wq[c*3+2]*xv[0][2][j];
    k[j] = wk[c*3+0]*xv[1][0][j] + wk[c*3+1]*xv[1][1][j] + wk[c*3+2]*xv[1][2][j];
    v[j] = wv[c*3+0]*xv[2][0][j] + wv[c*3+1]*xv[2][1][j] + wv[c*3+2]*xv[2][2][j];
    g[j] = wi[c*3+0]*xv[3][0][j] + wi[c*3+1]*xv[3][1][j] + wi[c*3+2]*xv[3][2][j];
  }
  float sq = q[0]*q[0] + q[1]*q[1] + q[2]*q[2] + q[3]*q[3];
  float sk = k[0]*k[0] + k[1]*k[1] + k[2]*k[2] + k[3]*k[3];
#pragma unroll
  for (int o = 16; o > 0; o >>= 1) {
    sq += __shfl_xor(sq, o, 32);
    sk += __shfl_xor(sk, o, 32);
  }
  const float rq = rsqrtf(sq * (1.f / HEAD_DIM) + 1e-5f);
  const float rk = rsqrtf(sk * (1.f / HEAD_DIM) + 1e-5f);
#pragma unroll
  for (int j = 0; j < 4; ++j) {
    q[j] *= rq * qnw[d0 + j];
    k[j] *= rk * knw[d0 + j];
  }
#pragma unroll
  for (int pp = 0; pp < 2; ++pp) {
    const int p = (d0 >> 1) + pp;
    const float c = fcos[t * (HEAD_DIM / 2) + p];
    const float s = fsin[t * (HEAD_DIM / 2) + p];
    const float qr = q[2*pp], qi = q[2*pp+1];
    q[2*pp]   = qr * c - qi * s;
    q[2*pp+1] = qr * s + qi * c;
    const float kr = k[2*pp], ki = k[2*pp+1];
    k[2*pp]   = kr * c - ki * s;
    k[2*pp+1] = kr * s + ki * c;
  }
  const float qscale = 0.08838834764831845f;  // 1/sqrt(128) folded into Q
  const size_t oidx = ((size_t)(b * N_HEAD + h) * SEQ + t) * HEAD_DIM + d0;
#pragma unroll
  for (int j = 0; j < 4; ++j) {
    qh[oidx + j]   = (bf16_t)(q[j] * qscale);
    kh[oidx + j]   = (bf16_t)k[j];
    vh[oidx + j]   = (bf16_t)v[j];
    gate[oidx + j] = 1.f / (1.f + __expf(-g[j]));
  }
}

// ---------------------------------------------------------------------------
// Flash attention, one wave per (16-query tile, head). Key tiles of 32.
// All 8 K-fragments loaded before the 8 score WMMAs (overlap on LOADcnt).
// ---------------------------------------------------------------------------
__global__ void __launch_bounds__(32)
flash_attn_wmma(const bf16_t* __restrict__ qh, const bf16_t* __restrict__ kh,
                const bf16_t* __restrict__ vh, const float* __restrict__ gate,
                bf16_t* __restrict__ y) {
  __shared__ float pbuf[16 * 32];
  const int lane = threadIdx.x;
  const int qt = blockIdx.x;
  const int bh = blockIdx.y;
  const int b = bh >> 4, h = bh & 15;
  const size_t hoff = (size_t)bh * SEQ * HEAD_DIM;
  const bf16_t* Q  = qh + hoff;
  const bf16_t* Kc = kh + hoff;
  const bf16_t* Vc = vh + hoff;
  const int n    = lane & 15;
  const int half = lane >> 4;

  bf16x16 qa[4];
  {
    const int kb = half ? 8 : 0;
    const bf16_t* p = Q + (size_t)(qt * 16 + n) * HEAD_DIM + kb;
#pragma unroll
    for (int kk = 0; kk < 4; ++kk) {
      B128c c0 = *(const B128c*)(p + kk * 32);
      B128c c1 = *(const B128c*)(p + kk * 32 + 16);
#pragma unroll
      for (int i = 0; i < 8; ++i) { qa[kk][i] = c0.h[i]; qa[kk][i + 8] = c1.h[i]; }
    }
  }

  f32x8 o[8];
  const f32x8 zero = {};
#pragma unroll
  for (int j = 0; j < 8; ++j) o[j] = zero;
  float mrow[8], lrow[8];
#pragma unroll
  for (int r = 0; r < 8; ++r) { mrow[r] = -1e30f; lrow[r] = 0.f; }

  const int tmax = qt * 16 + 15;
  const int ko = half ? 8 : 0;
  for (int kb32 = 0; kb32 <= tmax; kb32 += 32) {
    if (kb32 + 32 <= tmax)  // gfx1250 global_prefetch_b8 for next key tile
      __builtin_prefetch(Kc + (size_t)(kb32 + 32 + n) * HEAD_DIM + ko, 0, 1);
    // ---- hoisted K fragment loads (8 frags), then 8 WMMAs ----
    bf16x16 fb0[4], fb1[4];
#pragma unroll
    for (int kk = 0; kk < 4; ++kk) {
      const bf16_t* p0 = Kc + (size_t)(kb32 + n)      * HEAD_DIM + kk * 32 + ko;
      const bf16_t* p1 = Kc + (size_t)(kb32 + 16 + n) * HEAD_DIM + kk * 32 + ko;
      B128c a0 = *(const B128c*)p0, a1 = *(const B128c*)(p0 + 16);
      B128c c0 = *(const B128c*)p1, c1 = *(const B128c*)(p1 + 16);
#pragma unroll
      for (int i = 0; i < 8; ++i) {
        fb0[kk][i] = a0.h[i]; fb0[kk][i + 8] = a1.h[i];
        fb1[kk][i] = c0.h[i]; fb1[kk][i + 8] = c1.h[i];
      }
    }
    f32x8 s0 = zero, s1 = zero;
#pragma unroll
    for (int kk = 0; kk < 4; ++kk) {
      s0 = __builtin_amdgcn_wmma_f32_16x16x32_bf16(false, qa[kk], false, fb0[kk],
                                                   (short)0, s0, false, false);
      s1 = __builtin_amdgcn_wmma_f32_16x16x32_bf16(false, qa[kk], false, fb1[kk],
                                                   (short)0, s1, false, false);
    }
    // ---- causal mask + online softmax ----
    float al[8];
#pragma unroll
    for (int r = 0; r < 8; ++r) {
      const int trow = qt * 16 + half * 8 + r;
      if (kb32 + n > trow)      s0[r] = -1e30f;
      if (kb32 + 16 + n > trow) s1[r] = -1e30f;
      float mx = fmaxf(s0[r], s1[r]);
#pragma unroll
      for (int ofs = 8; ofs > 0; ofs >>= 1) mx = fmaxf(mx, __shfl_xor(mx, ofs, 32));
      const float mnew  = fmaxf(mrow[r], mx);
      const float alpha = __expf(mrow[r] - mnew);
      const float p0v = __expf(s0[r] - mnew);
      const float p1v = __expf(s1[r] - mnew);
      s0[r] = p0v; s1[r] = p1v;
      float rs = p0v + p1v;
#pragma unroll
      for (int ofs = 8; ofs > 0; ofs >>= 1) rs += __shfl_xor(rs, ofs, 32);
      lrow[r] = lrow[r] * alpha + rs;
      mrow[r] = mnew;
      al[r] = alpha;
    }
#pragma unroll
    for (int j = 0; j < 8; ++j)
#pragma unroll
      for (int r = 0; r < 8; ++r) o[j][r] *= al[r];

    // ---- transpose P (C-layout -> A-layout) through LDS ----
    __syncthreads();
#pragma unroll
    for (int r = 0; r < 8; ++r) {
      pbuf[(half * 8 + r) * 32 + n]      = s0[r];
      pbuf[(half * 8 + r) * 32 + 16 + n] = s1[r];
    }
    __syncthreads();
    bf16x16 pa;
    {
      const int kbase = half ? 8 : 0;
#pragma unroll
      for (int i = 0; i < 8; ++i) {
        pa[i]     = (bf16_t)pbuf[n * 32 + kbase + i];
        pa[i + 8] = (bf16_t)pbuf[n * 32 + kbase + 16 + i];
      }
    }
    // ---- O += P @ V ----
#pragma unroll
    for (int j = 0; j < 8; ++j) {
      bf16x16 vb;
      const bf16_t* vp = Vc + (size_t)(kb32 + ko) * HEAD_DIM + j * 16 + n;
#pragma unroll
      for (int i = 0; i < 16; ++i) vb[i] = vp[(size_t)(i < 8 ? i : i + 8) * HEAD_DIM];
      o[j] = __builtin_amdgcn_wmma_f32_16x16x32_bf16(false, pa, false, vb,
                                                     (short)0, o[j], false, false);
    }
  }
  // ---- epilogue: normalize, gate, store y[B*T, C] bf16 ----
#pragma unroll
  for (int j = 0; j < 8; ++j) {
#pragma unroll
    for (int r = 0; r < 8; ++r) {
      const int trow = qt * 16 + half * 8 + r;
      const float gv = gate[hoff + (size_t)trow * HEAD_DIM + j * 16 + n];
      const float val = o[j][r] * (1.f / lrow[r]) * gv;
      y[((size_t)(b * SEQ + trow)) * D_MODEL + h * HEAD_DIM + j * 16 + n] =
          (bf16_t)val;
    }
  }
}

// ---------------------------------------------------------------------------
// Host-side launcher
// ---------------------------------------------------------------------------
extern "C" void kernel_launch(void* const* d_in, const int* in_sizes, int n_in,
                              void* d_out, int out_size, void* d_ws,
                              size_t ws_size, hipStream_t stream) {
  (void)in_sizes; (void)n_in; (void)out_size; (void)ws_size;
  const float* x     = (const float*)d_in[0];
  const float* fcos  = (const float*)d_in[1];
  const float* fsin  = (const float*)d_in[2];
  const float* Wqkv  = (const float*)d_in[3];
  const float* wq    = (const float*)d_in[4];
  const float* wk    = (const float*)d_in[5];
  const float* wv    = (const float*)d_in[6];
  const float* wi    = (const float*)d_in[7];
  const float* qnw   = (const float*)d_in[8];
  const float* knw   = (const float*)d_in[9];
  const float* Wproj = (const float*)d_in[10];
  float* out = (float*)d_out;

  char* ws = (char*)d_ws;
  size_t off = 0;
  auto alloc = [&](size_t bytes) -> void* {
    void* p = ws + off;
    off += (bytes + 255) & ~(size_t)255;
    return p;
  };
  bf16_t* xb     = (bf16_t*)alloc((size_t)ROWS * D_MODEL * 2);
  bf16_t* wqkvb  = (bf16_t*)alloc((size_t)D_MODEL * QKV_N * 2);
  bf16_t* wprojb = (bf16_t*)alloc((size_t)D_MODEL * D_MODEL * 2);
  bf16_t* qkvb   = (bf16_t*)alloc((size_t)ROWS * QKV_N * 2);
  const size_t hsz = (size_t)BATCH * N_HEAD * SEQ * HEAD_DIM;
  bf16_t* qhb = (bf16_t*)alloc(hsz * 2);
  bf16_t* khb = (bf16_t*)alloc(hsz * 2);
  bf16_t* vhb = (bf16_t*)alloc(hsz * 2);
  float*  gb  = (float*)alloc(hsz * 4);
  bf16_t* yb  = (bf16_t*)alloc((size_t)ROWS * D_MODEL * 2);

  // 1) fp32 -> bf16
  f32_to_bf16<<<(ROWS * D_MODEL + 255) / 256, 256, 0, stream>>>(
      x, xb, ROWS * D_MODEL);
  f32_to_bf16<<<(D_MODEL * QKV_N + 255) / 256, 256, 0, stream>>>(
      Wqkv, wqkvb, D_MODEL * QKV_N);
  f32_to_bf16<<<(D_MODEL * D_MODEL + 255) / 256, 256, 0, stream>>>(
      Wproj, wprojb, D_MODEL * D_MODEL);

  // 2) qkv = x @ Wqkv   (4096 x 8192 x 2048), TDM-staged WMMA GEMM
  gemm_bf16_wmma_tdm<false><<<dim3(QKV_N / BN, ROWS / BM), 256, 0, stream>>>(
      xb, wqkvb, qkvb, ROWS, QKV_N, D_MODEL);

  // 3) conv + rmsnorm + rope + gate
  conv_norm_rope<<<(BATCH * SEQ * N_HEAD * 32) / 256, 256, 0, stream>>>(
      qkvb, wq, wk, wv, wi, qnw, knw, fcos, fsin, qhb, khb, vhb, gb);

  // 4) flash attention + gating -> y
  flash_attn_wmma<<<dim3(SEQ / 16, BATCH * N_HEAD), 32, 0, stream>>>(
      qhb, khb, vhb, gb, yb);

  // 5) out = y @ Wproj  (4096 x 2048 x 2048), fp32 store
  gemm_bf16_wmma_tdm<true><<<dim3(D_MODEL / BN, ROWS / BM), 256, 0, stream>>>(
      yb, wprojb, out, ROWS, D_MODEL, D_MODEL);
}
